// PointNetModule_14216341750523
// MI455X (gfx1250) — matrix operand order repaired and checked by reading, not compiled, not run on hardware
//
// PointNet++ forward for MI455X (gfx1250, wave32, WMMA).
// Heavy math = grouped GEMMs via v_wmma_f32_16x16x32_bf16 (bf16 A/B, f32 acc).
//  - grid = (Ntiles, Mslabs): fp32 activation matrix A streams from HBM once;
//    weights (small) are the re-read operand and live in the 192MB L2.
//  - B panel staged to LDS with global_load_async_to_lds_b128 (double buffer,
//    s_wait_asynccnt + barriers), consumed with ds_load_b128 -> 8x less L0 traffic.
//  - BN stats fused into GEMM epilogue (lane shuffle + global f32 atomics);
//    BN+ReLU fused into next GEMM's A-fragment load (fp32 -> bf16 on the fly).
#include <hip/hip_runtime.h>

typedef __bf16 bf16_t;
typedef __bf16 v16bf __attribute__((ext_vector_type(16)));
typedef __bf16 v8bf  __attribute__((ext_vector_type(8)));
typedef float  v8f   __attribute__((ext_vector_type(8)));

#define BATCH 4

// ---------------------------------------------------------------- utility
__global__ void pn_zero(float* __restrict__ p, int n) {
  int i = blockIdx.x * blockDim.x + threadIdx.x;
  if (i < n) p[i] = 0.f;
}

// fp32 [rows x K] -> bf16 [rows x Kp], zero padded
__global__ void pn_cast_pad(const float* __restrict__ src, bf16_t* __restrict__ dst,
                            int K, int Kp, long total) {
  long i = (long)blockIdx.x * blockDim.x + threadIdx.x;
  if (i >= total) return;
  int  c = (int)(i % Kp);
  long r = i / Kp;
  dst[i] = (c < K) ? (bf16_t)src[r * K + c] : (bf16_t)0.f;
}

// structures (B,3,N) -> xyz (B,N,3)
__global__ void pn_xyz(const float* __restrict__ st, float* __restrict__ xyz,
                       int Nn, int total) {
  int i = blockIdx.x * blockDim.x + threadIdx.x;
  if (i >= total) return;
  int b = i / Nn, n = i % Nn;
  for (int c = 0; c < 3; ++c)
    xyz[(long)i * 3 + c] = st[((long)b * 3 + c) * Nn + n];
}

// ------------------------------------------------- farthest point sampling
__global__ void pn_fps(const float* __restrict__ xyz, int Nn, int S, int* __restrict__ idx) {
  __shared__ float dist[1024];
  __shared__ float rv[1024];
  __shared__ int   ri[1024];
  __shared__ int   sfar;
  const int t = threadIdx.x, b = blockIdx.x;
  const float* X = xyz + (long)b * Nn * 3;
  const float px = X[t * 3 + 0], py = X[t * 3 + 1], pz = X[t * 3 + 2];
  dist[t] = 1e10f;
  if (t == 0) sfar = 0;
  __syncthreads();
  for (int s = 0; s < S; ++s) {
    const int far = sfar;
    if (t == 0) idx[b * S + s] = far;
    const float cx = X[far * 3 + 0], cy = X[far * 3 + 1], cz = X[far * 3 + 2];
    const float dx = px - cx, dy = py - cy, dz = pz - cz;
    float nd = fminf(dist[t], dx * dx + dy * dy + dz * dz);
    dist[t] = nd;
    rv[t] = nd; ri[t] = t;
    __syncthreads();
    for (int off = blockDim.x >> 1; off > 0; off >>= 1) {
      if (t < off) {
        float ov = rv[t + off]; int oi = ri[t + off];
        if (ov > rv[t] || (ov == rv[t] && oi < ri[t])) { rv[t] = ov; ri[t] = oi; }
      }
      __syncthreads();
    }
    if (t == 0) sfar = ri[0];
    __syncthreads();
  }
}

__global__ void pn_gather_xyz(const float* __restrict__ xyz, const int* __restrict__ fidx,
                              float* __restrict__ nxyz, int Nn, int S, int total) {
  int i = blockIdx.x * blockDim.x + threadIdx.x;
  if (i >= total) return;
  int b = i / S, j = fidx[i];
  for (int c = 0; c < 3; ++c) nxyz[(long)i * 3 + c] = xyz[((long)b * Nn + j) * 3 + c];
}

// ball query: first `ns` in-radius indices ascending, pad with first
__global__ void pn_qball(const float* __restrict__ xyz, const float* __restrict__ nxyz,
                         int* __restrict__ out, int Nn, int S, int ns, float r2, int total) {
  int i = blockIdx.x * blockDim.x + threadIdx.x;
  if (i >= total) return;
  int b = i / S;
  const float cx = nxyz[(long)i * 3], cy = nxyz[(long)i * 3 + 1], cz = nxyz[(long)i * 3 + 2];
  const float* X = xyz + (long)b * Nn * 3;
  int cnt = 0, first = -1;
  for (int j = 0; j < Nn && cnt < ns; ++j) {
    float dx = X[j * 3] - cx, dy = X[j * 3 + 1] - cy, dz = X[j * 3 + 2] - cz;
    if (dx * dx + dy * dy + dz * dz <= r2) {
      if (first < 0) first = j;
      out[(long)i * ns + cnt] = j;
      ++cnt;
    }
  }
  for (; cnt < ns; ++cnt) out[(long)i * ns + cnt] = first;
}

// build grouped A matrix in bf16: rows (b,s,k), cols [xyz-center | pts | pad]
__global__ void pn_group(const float* __restrict__ xyz, const float* __restrict__ center,
                         const int* __restrict__ idx, const float* __restrict__ pts,
                         bf16_t* __restrict__ A, int S, int Kk, int Nn, int C, int Kp,
                         long total) {
  long i = (long)blockIdx.x * blockDim.x + threadIdx.x;
  if (i >= total) return;
  int  c  = (int)(i % Kp);
  long rw = i / Kp;
  int  k  = (int)(rw % Kk);
  long bs = rw / Kk;
  int  s  = (int)(bs % S);
  int  b  = (int)(bs / S);
  int  j  = idx ? idx[((long)b * S + s) * Kk + k] : k;
  float v = 0.f;
  if (c < 3) {
    v = xyz[((long)b * Nn + j) * 3 + c];
    if (center) v -= center[((long)b * S + s) * 3 + c];
  } else if (c < 3 + C) {
    v = pts[((long)b * C + (c - 3)) * Nn + j];
  }
  A[i] = (bf16_t)v;
}

// ------------------------------------------------------------------ GEMM
// Out[M x N] = A[M x kp] * W[N x kp]^T + bias; kp multiple of 128.
// grid(x = N/64, y = M/128); wave -> 16x64 tile, block -> 128x64.
// AMODE 0: A bf16.  AMODE 1: A fp32, apply relu(scale*x+shift) per K-column.
__device__ inline void pn_epilogue(v8f acc, int row0, int hl, int col, int ldo,
                                   const float* __restrict__ bias, float* __restrict__ Out,
                                   float* __restrict__ ssum, float* __restrict__ ssq,
                                   int do_stats) {
  const float bv = bias[col];
  float s1 = 0.f, s2 = 0.f;
#pragma unroll
  for (int i = 0; i < 8; ++i) {
    float v = acc[i] + bv;
    Out[(long)(row0 + hl * 8 + i) * ldo + col] = v;
    s1 += v; s2 += v * v;
  }
  if (do_stats) {
    s1 += __shfl_xor(s1, 16, 32);
    s2 += __shfl_xor(s2, 16, 32);
    if (hl == 0) { atomicAdd(ssum + col, s1); atomicAdd(ssq + col, s2); }
  }
}

template <int AMODE>
__global__ __launch_bounds__(256) void pn_gemm(
    const void* __restrict__ Abuf, const bf16_t* __restrict__ W, int kp,
    const float* __restrict__ bias, const float* __restrict__ scale,
    const float* __restrict__ shift, float* __restrict__ Out, int N,
    float* __restrict__ ssum, float* __restrict__ ssq, int do_stats) {
  // double-buffered B panel: 64 cols x 128 K of bf16 = 16KB per buffer
  __shared__ __align__(32) bf16_t Bsh[2][64 * 128];

  const int lane = threadIdx.x & 31;
  const int wave = threadIdx.x >> 5;
  const int col0 = blockIdx.x * 64;
  const int row0 = blockIdx.y * 128 + wave * 16;
  const int rl = lane & 15;   // A row / B col within tile
  const int hl = lane >> 4;   // lane half
  const long arow = (long)(row0 + rl);
  const int NC = kp >> 7;     // 128-K chunks

  // async-stage one 64x128 B chunk into Bsh[buf]; wave w covers panel rows 8w..8w+7
  auto stage = [&](int chunk, int buf) {
    const int kc = chunk << 7;
#pragma unroll
    for (int i = 0; i < 4; ++i) {
      const int row = wave * 8 + i * 2 + hl;              // 2 rows per async op
      const bf16_t* g = W + (size_t)(col0 + row) * kp + kc + rl * 8;
      unsigned l = (unsigned)(uintptr_t)(&Bsh[buf][row * 128 + rl * 8]);
      asm volatile("global_load_async_to_lds_b128 %0, %1, off"
                   :: "v"(l), "v"(g) : "memory");
    }
  };

  v8f acc0 = {}, acc1 = {}, acc2 = {}, acc3 = {};

  stage(0, 0);
  for (int c = 0; c < NC; ++c) {
    const int buf = c & 1;
    if (c + 1 < NC) {
      stage(c + 1, buf ^ 1);
      asm volatile("s_wait_asynccnt 0x4" ::: "memory");   // chunk c landed
    } else {
      asm volatile("s_wait_asynccnt 0x0" ::: "memory");
    }
    __syncthreads();
    const bf16_t* bb = &Bsh[buf][0];
    const int kcb = c << 7;
    // prefetch next A chunk for this lane's row
    if (kcb + 128 < kp) {
      if (AMODE == 0)
        __builtin_prefetch((const bf16_t*)Abuf + arow * kp + kcb + 128 + hl * 8, 0, 0);
      else
        __builtin_prefetch((const float*)Abuf + arow * kp + kcb + 128 + hl * 8, 0, 0);
    }
#pragma unroll
    for (int ks = 0; ks < 4; ++ks) {
      const int k0 = kcb + ks * 32;
      // ---- A fragment: lane half hl covers K = kb..kb+7 and kb+16..kb+23
      const int kb = k0 + hl * 8;
      v16bf a;
      if (AMODE == 0) {
        const bf16_t* Ap = (const bf16_t*)Abuf + arow * kp + kb;
        v8bf lo = *(const v8bf*)Ap;
        v8bf hi = *(const v8bf*)(Ap + 16);
#pragma unroll
        for (int i = 0; i < 8; ++i) { a[i] = lo[i]; a[8 + i] = hi[i]; }
      } else {
        const float* Ap = (const float*)Abuf + arow * kp + kb;
#pragma unroll
        for (int i = 0; i < 8; ++i) {
          float v0 = fmaxf(Ap[i]      * scale[kb + i]      + shift[kb + i],      0.f);
          float v1 = fmaxf(Ap[16 + i] * scale[kb + 16 + i] + shift[kb + 16 + i], 0.f);
          a[i]     = (bf16_t)v0;
          a[8 + i] = (bf16_t)v1;
        }
      }
      // ---- B fragments from LDS: lane = panel row (t*16+rl), K = kof..kof+7 and kof+16..kof+23
      const int kof = ks * 32 + hl * 8;
      v16bf b0, b1, b2, b3;
      {
        v8bf l0 = *(const v8bf*)(bb + ( 0 + rl) * 128 + kof), h0 = *(const v8bf*)(bb + ( 0 + rl) * 128 + kof + 16);
        v8bf l1 = *(const v8bf*)(bb + (16 + rl) * 128 + kof), h1 = *(const v8bf*)(bb + (16 + rl) * 128 + kof + 16);
        v8bf l2 = *(const v8bf*)(bb + (32 + rl) * 128 + kof), h2 = *(const v8bf*)(bb + (32 + rl) * 128 + kof + 16);
        v8bf l3 = *(const v8bf*)(bb + (48 + rl) * 128 + kof), h3 = *(const v8bf*)(bb + (48 + rl) * 128 + kof + 16);
#pragma unroll
        for (int i = 0; i < 8; ++i) {
          b0[i] = l0[i]; b0[8 + i] = h0[i];
          b1[i] = l1[i]; b1[8 + i] = h1[i];
          b2[i] = l2[i]; b2[8 + i] = h2[i];
          b3[i] = l3[i]; b3[8 + i] = h3[i];
        }
      }
      acc0 = __builtin_amdgcn_wmma_f32_16x16x32_bf16(false, a, false, b0, (short)0, acc0, false, false);
      acc1 = __builtin_amdgcn_wmma_f32_16x16x32_bf16(false, a, false, b1, (short)0, acc1, false, false);
      acc2 = __builtin_amdgcn_wmma_f32_16x16x32_bf16(false, a, false, b2, (short)0, acc2, false, false);
      acc3 = __builtin_amdgcn_wmma_f32_16x16x32_bf16(false, a, false, b3, (short)0, acc3, false, false);
    }
    __syncthreads();  // protect Bsh[buf] before it is overwritten
  }
  pn_epilogue(acc0, row0, hl, col0 +  0 + rl, N, bias, Out, ssum, ssq, do_stats);
  pn_epilogue(acc1, row0, hl, col0 + 16 + rl, N, bias, Out, ssum, ssq, do_stats);
  pn_epilogue(acc2, row0, hl, col0 + 32 + rl, N, bias, Out, ssum, ssq, do_stats);
  pn_epilogue(acc3, row0, hl, col0 + 48 + rl, N, bias, Out, ssum, ssq, do_stats);
}

// scale/shift from accumulated stats: bn(x) = scale*x + shift
__global__ void pn_bnfin(const float* __restrict__ ssum, const float* __restrict__ ssq,
                         const float* __restrict__ g, const float* __restrict__ be,
                         float* __restrict__ scl, float* __restrict__ sft, int n, float invM) {
  int i = blockIdx.x * blockDim.x + threadIdx.x;
  if (i >= n) return;
  float m = ssum[i] * invM;
  float var = ssq[i] * invM - m * m;
  float s = g[i] * rsqrtf(var + 1e-5f);
  scl[i] = s;
  sft[i] = be[i] - m * s;
}

// out[b,s,c] = relu( max_k ( scale[c]*X[(b,s,k),c]+shift[c] ) )
__global__ void pn_bnrelu_maxpool(const float* __restrict__ X, const float* __restrict__ scl,
                                  const float* __restrict__ sft, float* __restrict__ out,
                                  int Kk, int C, int total) {
  int i = blockIdx.x * blockDim.x + threadIdx.x;
  if (i >= total) return;
  int c = i % C;
  long bs = i / C;
  const float s = scl[c], t = sft[c];
  float m = -3.4e38f;
  for (int k = 0; k < Kk; ++k) m = fmaxf(m, X[((long)bs * Kk + k) * C + c] * s + t);
  out[i] = fmaxf(m, 0.f);
}

// comb[(b*640+c), col] : col<S -> pool[b,s=col,c] ; else lin[(b*640+c), col-S]
__global__ void pn_build_comb(const float* __restrict__ pool, const float* __restrict__ lin,
                              bf16_t* __restrict__ comb, int S, int C, long total) {
  long i = (long)blockIdx.x * blockDim.x + threadIdx.x;
  if (i >= total) return;
  int  col = (int)(i % (2 * S));
  long r   = i / (2 * S);
  int  c   = (int)(r % C);
  int  b   = (int)(r / C);
  float v = (col < S) ? pool[((long)b * S + col) * C + c] : lin[r * S + (col - S)];
  comb[i] = (bf16_t)v;
}

// ----------------------------------------------- tail MLP (4 rows, tiny)
__global__ __launch_bounds__(256) void pn_tail(
    const float* __restrict__ feat,
    const float* __restrict__ w1, const float* __restrict__ b1,
    const float* __restrict__ g1, const float* __restrict__ be1,
    const float* __restrict__ w2, const float* __restrict__ b2,
    const float* __restrict__ g2, const float* __restrict__ be2,
    const float* __restrict__ w3, const float* __restrict__ b3,
    float* __restrict__ out) {
  __shared__ float sf[4 * 640];
  __shared__ float h1[4 * 512];
  __shared__ float h2[4 * 256];
  const int t = threadIdx.x;
  for (int i = t; i < 4 * 640; i += 256) sf[i] = feat[i];
  __syncthreads();
  for (int i = t; i < 4 * 512; i += 256) {
    int r = i >> 9, o = i & 511;
    float s = b1[o];
    const float* w = w1 + (long)o * 640;
    const float* x = sf + r * 640;
    for (int k = 0; k < 640; ++k) s += x[k] * w[k];
    h1[i] = s;
  }
  __syncthreads();
  for (int o = t; o < 512; o += 256) {
    float m = 0.f; for (int r = 0; r < 4; ++r) m += h1[r * 512 + o]; m *= 0.25f;
    float v = 0.f; for (int r = 0; r < 4; ++r) { float d = h1[r * 512 + o] - m; v += d * d; } v *= 0.25f;
    float sc = g1[o] * rsqrtf(v + 1e-5f);
    for (int r = 0; r < 4; ++r) h1[r * 512 + o] = fmaxf((h1[r * 512 + o] - m) * sc + be1[o], 0.f);
  }
  __syncthreads();
  for (int i = t; i < 4 * 256; i += 256) {
    int r = i >> 8, o = i & 255;
    float s = b2[o];
    const float* w = w2 + (long)o * 512;
    const float* x = h1 + r * 512;
    for (int k = 0; k < 512; ++k) s += x[k] * w[k];
    h2[i] = s;
  }
  __syncthreads();
  if (t < 256) {
    int o = t;
    float m = 0.f; for (int r = 0; r < 4; ++r) m += h2[r * 256 + o]; m *= 0.25f;
    float v = 0.f; for (int r = 0; r < 4; ++r) { float d = h2[r * 256 + o] - m; v += d * d; } v *= 0.25f;
    float sc = g2[o] * rsqrtf(v + 1e-5f);
    for (int r = 0; r < 4; ++r) h2[r * 256 + o] = fmaxf((h2[r * 256 + o] - m) * sc + be2[o], 0.f);
  }
  __syncthreads();
  for (int i = t; i < 40; i += 256) {
    int r = i / 10, o = i % 10;
    float s = b3[o];
    const float* w = w3 + o * 256;
    const float* x = h2 + r * 256;
    for (int k = 0; k < 256; ++k) s += x[k] * w[k];
    out[i] = s;
  }
}

// =================================================================== host
extern "C" void kernel_launch(void* const* d_in, const int* in_sizes, int n_in,
                              void* d_out, int out_size, void* d_ws, size_t ws_size,
                              hipStream_t stream) {
  (void)in_sizes; (void)n_in; (void)out_size; (void)ws_size;
  int p = 0;
  const float* structures = (const float*)d_in[p++];
  const float* e9  = (const float*)d_in[p++];
  const float* e19 = (const float*)d_in[p++];
  const float* e29 = (const float*)d_in[p++];
  struct MlpL { const float *W, *b, *g, *be; };
  MlpL sa[3][3];
  for (int m = 0; m < 3; ++m)
    for (int l = 0; l < 3; ++l) {
      sa[m][l].W  = (const float*)d_in[p++];
      sa[m][l].b  = (const float*)d_in[p++];
      sa[m][l].g  = (const float*)d_in[p++];
      sa[m][l].be = (const float*)d_in[p++];
    }
  const float* sa1l1W = (const float*)d_in[p++]; const float* sa1l1b = (const float*)d_in[p++];
  const float* sa1l2W = (const float*)d_in[p++]; const float* sa1l2b = (const float*)d_in[p++];
  const float* sa2l1W = (const float*)d_in[p++]; const float* sa2l1b = (const float*)d_in[p++];
  const float* sa2l2W = (const float*)d_in[p++]; const float* sa2l2b = (const float*)d_in[p++];
  const float* fc1W = (const float*)d_in[p++]; const float* fc1b = (const float*)d_in[p++];
  const float* bn1g = (const float*)d_in[p++]; const float* bn1b = (const float*)d_in[p++];
  const float* fc2W = (const float*)d_in[p++]; const float* fc2b = (const float*)d_in[p++];
  const float* bn2g = (const float*)d_in[p++]; const float* bn2b = (const float*)d_in[p++];
  const float* fc3W = (const float*)d_in[p++]; const float* fc3b = (const float*)d_in[p++];

  // ---------------- workspace layout (deterministic)
  char* base = (char*)d_ws;
  size_t off = 0;
  auto alloc = [&](size_t bytes) -> char* {
    char* r = base + off;
    off = (off + bytes + 255) & ~(size_t)255;
    return r;
  };
  // layer-0 K padded 643 -> 768 so every GEMM K is a multiple of the 128-K LDS chunk
  const int wrows[3] = {640, 1280, 640};
  const int wkin[3]  = {643, 640, 1280};
  const int wkp[3]   = {768, 640, 1280};
  bf16_t* saWb[3][3];
  for (int m = 0; m < 3; ++m)
    for (int l = 0; l < 3; ++l)
      saWb[m][l] = (bf16_t*)alloc((size_t)wrows[l] * wkp[l] * 2);
  bf16_t* wl19 = (bf16_t*)alloc((size_t)512 * 1024 * 2);
  bf16_t* wl12 = (bf16_t*)alloc((size_t)512 * 1024 * 2);
  bf16_t* wl29 = (bf16_t*)alloc((size_t)128 * 1024 * 2);
  bf16_t* wl22 = (bf16_t*)alloc((size_t)128 * 256 * 2);
  float* xyz0  = (float*)alloc((size_t)BATCH * 1024 * 3 * 4);
  int*   fidx1 = (int*)alloc((size_t)BATCH * 512 * 4);
  float* nxyz1 = (float*)alloc((size_t)BATCH * 512 * 3 * 4);
  int*   bidx1 = (int*)alloc((size_t)BATCH * 512 * 32 * 4);
  int*   fidx2 = (int*)alloc((size_t)BATCH * 128 * 4);
  float* nxyz2 = (float*)alloc((size_t)BATCH * 128 * 3 * 4);
  int*   bidx2 = (int*)alloc((size_t)BATCH * 128 * 64 * 4);
  bf16_t* A0 = (bf16_t*)alloc((size_t)65536 * 768 * 2);
  float*  XA = (float*)alloc((size_t)65536 * 640 * 4);
  float*  XB = (float*)alloc((size_t)65536 * 1280 * 4);
  float* ssum = (float*)alloc(1280 * 4);
  float* ssq  = (float*)alloc(1280 * 4);
  float* scl  = (float*)alloc(1280 * 4);
  float* sft  = (float*)alloc(1280 * 4);
  float* pool1 = (float*)alloc((size_t)BATCH * 512 * 640 * 4);
  float* pool2 = (float*)alloc((size_t)BATCH * 128 * 640 * 4);
  float* pool3 = (float*)alloc((size_t)BATCH * 640 * 4);
  bf16_t* e19b  = (bf16_t*)alloc((size_t)2560 * 1024 * 2);
  float*  lin19 = (float*)alloc((size_t)2560 * 512 * 4);
  bf16_t* comb1 = (bf16_t*)alloc((size_t)2560 * 1024 * 2);
  float*  l1    = (float*)alloc((size_t)2560 * 512 * 4);
  bf16_t* e29b  = (bf16_t*)alloc((size_t)2560 * 1024 * 2);
  float*  lin29 = (float*)alloc((size_t)2560 * 128 * 4);
  bf16_t* comb2 = (bf16_t*)alloc((size_t)2560 * 256 * 2);
  float*  l2    = (float*)alloc((size_t)2560 * 128 * 4);

  auto castpad = [&](const float* s, bf16_t* d, int K, int Kp, long rows) {
    long total = rows * Kp;
    pn_cast_pad<<<dim3((unsigned)((total + 255) / 256)), 256, 0, stream>>>(s, d, K, Kp, total);
  };
  auto zero_stats = [&]() { pn_zero<<<(2560 + 255) / 256, 256, 0, stream>>>(ssum, 2560); };
  auto gemm = [&](const void* A, int amode, const bf16_t* Wb, int kp, const float* bias,
                  float* out, int M, int N, int stats) {
    dim3 g(N / 64, M / 128);   // x = col tiles (fast), y = row slabs -> A streams once
    if (amode == 0)
      pn_gemm<0><<<g, 256, 0, stream>>>(A, Wb, kp, bias, scl, sft, out, N, ssum, ssq, stats);
    else
      pn_gemm<1><<<g, 256, 0, stream>>>(A, Wb, kp, bias, scl, sft, out, N, ssum, ssq, stats);
  };
  auto bnfin = [&](int N, const float* g, const float* be, int M) {
    pn_bnfin<<<(N + 255) / 256, 256, 0, stream>>>(ssum, ssq, g, be, scl, sft, N, 1.f / (float)M);
  };
  auto run_sa = [&](int M, const MlpL* mlp, bf16_t* const* Wb, float* pool, int S_, int Kk_) {
    zero_stats();
    gemm(A0, 0, Wb[0], 768, mlp[0].b, XA, M, 640, 1);
    bnfin(640, mlp[0].g, mlp[0].be, M);
    zero_stats();
    gemm(XA, 1, Wb[1], 640, mlp[1].b, XB, M, 1280, 1);
    bnfin(1280, mlp[1].g, mlp[1].be, M);
    zero_stats();
    gemm(XB, 1, Wb[2], 1280, mlp[2].b, XA, M, 640, 1);
    bnfin(640, mlp[2].g, mlp[2].be, M);
    int total = BATCH * S_ * 640;
    pn_bnrelu_maxpool<<<(total + 255) / 256, 256, 0, stream>>>(XA, scl, sft, pool, Kk_, 640, total);
  };

  // -------- weight + embedding conversion to padded bf16
  for (int m = 0; m < 3; ++m)
    for (int l = 0; l < 3; ++l)
      castpad(sa[m][l].W, saWb[m][l], wkin[l], wkp[l], wrows[l]);
  castpad(sa1l1W, wl19, 1000, 1024, 512);
  castpad(sa1l2W, wl12, 1024, 1024, 512);
  castpad(sa2l1W, wl29, 1000, 1024, 128);
  castpad(sa2l2W, wl22, 256, 256, 128);
  castpad(e19, e19b, 1000, 1024, 2560);
  castpad(e29, e29b, 1000, 1024, 2560);

  // -------- level 1: sample 512, group 32, mlp, pool, combine with e19
  pn_xyz<<<(BATCH * 1024 + 255) / 256, 256, 0, stream>>>(structures, xyz0, 1024, BATCH * 1024);
  pn_fps<<<BATCH, 1024, 0, stream>>>(xyz0, 1024, 512, fidx1);
  pn_gather_xyz<<<(BATCH * 512 + 255) / 256, 256, 0, stream>>>(xyz0, fidx1, nxyz1, 1024, 512, BATCH * 512);
  pn_qball<<<(BATCH * 512 + 255) / 256, 256, 0, stream>>>(xyz0, nxyz1, bidx1, 1024, 512, 32, 0.04f, BATCH * 512);
  {
    long total = (long)BATCH * 512 * 32 * 768;
    pn_group<<<dim3((unsigned)((total + 255) / 256)), 256, 0, stream>>>(
        xyz0, nxyz1, bidx1, e9, A0, 512, 32, 1024, 640, 768, total);
  }
  run_sa(65536, sa[0], saWb[0], pool1, 512, 32);
  gemm(e19b, 0, wl19, 1024, sa1l1b, lin19, 2560, 512, 0);
  {
    long total = (long)2560 * 1024;
    pn_build_comb<<<dim3((unsigned)((total + 255) / 256)), 256, 0, stream>>>(pool1, lin19, comb1, 512, 640, total);
  }
  gemm(comb1, 0, wl12, 1024, sa1l2b, l1, 2560, 512, 0);  // l1: (B,640,512) chan-major

  // -------- level 2: sample 128, group 64, mlp, pool, combine with e29
  pn_fps<<<BATCH, 512, 0, stream>>>(nxyz1, 512, 128, fidx2);
  pn_gather_xyz<<<(BATCH * 128 + 255) / 256, 256, 0, stream>>>(nxyz1, fidx2, nxyz2, 512, 128, BATCH * 128);
  pn_qball<<<(BATCH * 128 + 255) / 256, 256, 0, stream>>>(nxyz1, nxyz2, bidx2, 512, 128, 64, 0.16f, BATCH * 128);
  {
    long total = (long)BATCH * 128 * 64 * 768;
    pn_group<<<dim3((unsigned)((total + 255) / 256)), 256, 0, stream>>>(
        nxyz1, nxyz2, bidx2, l1, A0, 128, 64, 512, 640, 768, total);
  }
  run_sa(32768, sa[1], saWb[1], pool2, 128, 64);
  gemm(e29b, 0, wl29, 1024, sa2l1b, lin29, 2560, 128, 0);
  {
    long total = (long)2560 * 256;
    pn_build_comb<<<dim3((unsigned)((total + 255) / 256)), 256, 0, stream>>>(pool2, lin29, comb2, 128, 640, total);
  }
  gemm(comb2, 0, wl22, 256, sa2l2b, l2, 2560, 128, 0);   // l2: (B,640,128) chan-major

  // -------- level 3: group_all (no centering, identity idx), mlp, pool
  {
    long total = (long)BATCH * 128 * 768;
    pn_group<<<dim3((unsigned)((total + 255) / 256)), 256, 0, stream>>>(
        nxyz2, nullptr, nullptr, l2, A0, 1, 128, 128, 640, 768, total);
  }
  run_sa(512, sa[2], saWb[2], pool3, 1, 128);

  // -------- classifier head
  pn_tail<<<1, 256, 0, stream>>>(pool3, fc1W, fc1b, bn1g, bn1b, fc2W, fc2b, bn2g, bn2b,
                                 fc3W, fc3b, (float*)d_out);
}
